// MinimalGPT_7164005450174
// MI455X (gfx1250) — matrix operand order repaired and checked by reading, not compiled, not run on hardware
//
#include <hip/hip_runtime.h>
#include <cstdint>
#include <cstddef>

// ---------------------------------------------------------------------------
// Types for CDNA5 WMMA (wave32, 16x16x32 bf16 -> f32) and TDM descriptors
// ---------------------------------------------------------------------------
typedef __bf16 bf16_t;
typedef bf16_t bf16x16 __attribute__((ext_vector_type(16)));
typedef float  f32x8   __attribute__((ext_vector_type(8)));
typedef unsigned u32x4 __attribute__((ext_vector_type(4)));
typedef int      i32x8 __attribute__((ext_vector_type(8)));
typedef int      i32x4 __attribute__((ext_vector_type(4)));

union FragAB { bf16x16 v; unsigned u[8]; };

__device__ __forceinline__ f32x8 wmma_bf16(const FragAB& a, const FragAB& b, f32x8 c) {
    return __builtin_amdgcn_wmma_f32_16x16x32_bf16(
        /*neg_a=*/false, a.v, /*neg_b=*/false, b.v,
        /*c_mod=*/(short)0, c, /*reuse_a=*/false, /*reuse_b=*/false);
}

// ---------------------------------------------------------------------------
// Tensor Data Mover: async 2-D bf16 tile load global -> LDS.
// D# packing per CDNA5 ISA 8.3/8.4 (group0: count/lds_addr/global_addr/type,
// group1: data_size + dims + tile dims + dim0 stride [+ LDS pad control]).
// Flat __shared__ pointers truncate to the LDS byte offset (ISA 10.2 aperture
// mapping discards upper bits), so (unsigned)(uintptr_t)p is the lds_addr.
// This toolchain exposes the 6-arg builtin: (g0, g1, g2, g3, g_extra, cpol).
// ---------------------------------------------------------------------------
__device__ __forceinline__ void tdm_load_tile_2d(
    unsigned lds_off, const bf16_t* gsrc,
    unsigned tensor_d0, unsigned tensor_d1, unsigned stride0_elems,
    unsigned tile_d0, unsigned tile_d1, unsigned pad_bits) {
    unsigned long long ga = (unsigned long long)(uintptr_t)gsrc;
    u32x4 g0;
    g0[0] = 1u;                                          // count=1, user descriptor
    g0[1] = lds_off;                                     // LDS byte address
    g0[2] = (unsigned)(ga & 0xFFFFFFFFull);              // global_addr[31:0]
    g0[3] = (unsigned)((ga >> 32) & 0x01FFFFFFull)       // global_addr[56:32]
            | (2u << 30);                                // type = 2 ("image")
    i32x8 g1;
    g1[0] = (int)((1u << 16) | pad_bits);                // data_size=1 (2B) + pad ctl
    g1[1] = (int)((tensor_d0 & 0xFFFFu) << 16);          // tensor_dim0[15:0]
    g1[2] = (int)(((tensor_d0 >> 16) & 0xFFFFu) |
                  ((tensor_d1 & 0xFFFFu) << 16));        // td0[31:16] | td1[15:0]
    g1[3] = (int)(((tensor_d1 >> 16) & 0xFFFFu) |
                  (tile_d0 << 16));                      // td1[31:16] | tile_dim0
    g1[4] = (int)(tile_d1 & 0xFFFFu);                    // tile_dim1 (tile_dim2=0)
    g1[5] = (int)stride0_elems;                          // tensor_dim0_stride[31:0]
    g1[6] = 0;                                           // stride0[47:32] | stride1 lo
    g1[7] = 0;
    i32x4 z4 = {0, 0, 0, 0};                             // groups 2/3 unused (2-D)
    i32x8 z8 = {0, 0, 0, 0, 0, 0, 0, 0};
    __builtin_amdgcn_tensor_load_to_lds(g0, g1, z4, z4, z8, 0);
}

// pad control for A tiles: insert 16B pad every 64B => LDS row stride 40 bf16.
// pad_enable(bit20) | pad_interval=3 (64B, bits24:22) | pad_amount=3 (4 dwords, bits31:25)
#define TDM_PAD_40 ((1u << 20) | (3u << 22) | (3u << 25))

// ---------------------------------------------------------------------------
// fp32 -> bf16 conversion (weights, one shot per layer)
// ---------------------------------------------------------------------------
__global__ __launch_bounds__(256) void f2bf_kernel(const float* __restrict__ src,
                                                   bf16_t* __restrict__ dst, size_t n) {
    size_t i = (size_t)blockIdx.x * 256u + threadIdx.x;
    if (i < n) dst[i] = (bf16_t)src[i];
}

// ---------------------------------------------------------------------------
// Embedding gather: h[s][d] = emb[tokens[s]][d]
// ---------------------------------------------------------------------------
__global__ __launch_bounds__(256) void embed_kernel(const int* __restrict__ tok,
                                                    const float* __restrict__ emb,
                                                    float* __restrict__ h, int S, int D) {
    size_t i = (size_t)blockIdx.x * 256u + threadIdx.x;
    if (i < (size_t)S * D) {
        int s = (int)(i / D), d = (int)(i % D);
        h[i] = emb[(size_t)tok[s] * D + d];
    }
}

// ---------------------------------------------------------------------------
// RMSNorm: one block per row, bf16 output (feeds WMMA GEMMs)
// ---------------------------------------------------------------------------
__global__ __launch_bounds__(256) void rmsnorm_bf16_kernel(const float* __restrict__ x,
                                                           const float* __restrict__ w,
                                                           bf16_t* __restrict__ out, int D) {
    int s = blockIdx.x;
    const float* row = x + (size_t)s * D;
    float ss = 0.f;
    for (int i = threadIdx.x; i < D; i += 256) { float v = row[i]; ss += v * v; }
    __shared__ float red[8];
    __shared__ float tot;
    for (int o = 16; o > 0; o >>= 1) ss += __shfl_down(ss, o, 32);
    if ((threadIdx.x & 31) == 0) red[threadIdx.x >> 5] = ss;
    __syncthreads();
    if (threadIdx.x == 0) {
        float t = 0.f;
        for (int i = 0; i < 8; i++) t += red[i];
        tot = rsqrtf(t / (float)D + 1e-6f);
    }
    __syncthreads();
    float rs = tot;
    for (int i = threadIdx.x; i < D; i += 256)
        out[(size_t)s * D + i] = (bf16_t)(row[i] * rs * w[i]);
}

// Final-row RMSNorm, f32 output (feeds logits GEMV)
__global__ __launch_bounds__(256) void final_norm_kernel(const float* __restrict__ x,
                                                         const float* __restrict__ w,
                                                         float* __restrict__ out,
                                                         int D, int row) {
    const float* r = x + (size_t)row * D;
    float ss = 0.f;
    for (int i = threadIdx.x; i < D; i += 256) { float v = r[i]; ss += v * v; }
    __shared__ float red[8];
    __shared__ float tot;
    for (int o = 16; o > 0; o >>= 1) ss += __shfl_down(ss, o, 32);
    if ((threadIdx.x & 31) == 0) red[threadIdx.x >> 5] = ss;
    __syncthreads();
    if (threadIdx.x == 0) {
        float t = 0.f;
        for (int i = 0; i < 8; i++) t += red[i];
        tot = rsqrtf(t / (float)D + 1e-6f);
    }
    __syncthreads();
    float rs = tot;
    for (int i = threadIdx.x; i < D; i += 256) out[i] = r[i] * rs * w[i];
}

// ---------------------------------------------------------------------------
// RoPE: fp32 in, bf16 out, same [S][D] layout (interleaved pairs per head)
// ---------------------------------------------------------------------------
__global__ __launch_bounds__(256) void rope_bf16_kernel(const float* __restrict__ x,
                                                        bf16_t* __restrict__ out,
                                                        int S, int D, int hd) {
    size_t p = (size_t)blockIdx.x * 256u + threadIdx.x;
    size_t np = (size_t)S * D / 2;
    if (p >= np) return;
    int s  = (int)(p / (D / 2));
    int wi = (int)(p % (D / 2));
    int h  = wi / (hd / 2);
    int i  = wi % (hd / 2);
    int c0 = h * hd + 2 * i;
    float fr = powf(10000.f, -(float)(2 * i) / (float)hd);
    float ang = (float)s * fr;
    float cs = cosf(ang), sn = sinf(ang);
    float x0 = x[(size_t)s * D + c0];
    float x1 = x[(size_t)s * D + c0 + 1];
    out[(size_t)s * D + c0]     = (bf16_t)(x0 * cs - x1 * sn);
    out[(size_t)s * D + c0 + 1] = (bf16_t)(x0 * sn + x1 * cs);
}

// V transpose: f32 [S][D] -> bf16 [D][S] (row c = h*64+d), so P x V B-fragments
// get contiguous key-pairs.
__global__ __launch_bounds__(256) void vtrans_kernel(const float* __restrict__ v,
                                                     bf16_t* __restrict__ vt, int S, int D) {
    size_t i = (size_t)blockIdx.x * 256u + threadIdx.x;
    if (i < (size_t)S * D) {
        int s = (int)(i / D), c = (int)(i % D);
        vt[(size_t)c * S + s] = (bf16_t)v[i];
    }
}

// ---------------------------------------------------------------------------
// Tiled bf16 WMMA GEMM with TDM double buffering.
//   C(MxN) = A(MxK,row-major bf16) * B(KxN,row-major bf16)
// Block tile 128x128, K-step 32, 8 waves; wave tile 32x64 (2x4 WMMA tiles).
// Wave 0 posts two TDM 2-D tile descriptors per K-step (A with DMA-side LDS
// padding to a 40-element row stride; B row-major 32x128), waits TENSORcnt,
// then a workgroup barrier publishes the LDS tile to all waves while the
// next tile streams in behind it.
// EPI: 0 = f32 store, 1 = f32 +=, 2 = bf16( silu(Aux) * acc )
// ---------------------------------------------------------------------------
template <int EPI>
__global__ __launch_bounds__(256) void gemm_bf16_kernel(
    const bf16_t* __restrict__ A, const bf16_t* __restrict__ B,
    void* __restrict__ OutV, const float* __restrict__ Aux,
    int M, int N, int K) {
    constexpr int BM = 128, BN = 128, BK = 32, LSTA = 40;
    __shared__ bf16_t As[2][BM * LSTA];   // padded rows (pad made by TDM)
    __shared__ bf16_t Bs[2][BK * BN];     // row-major [k][n]

    const int tid  = threadIdx.x;
    const int lane = tid & 31;
    const int wave = tid >> 5;
    const int wm   = wave >> 1;   // 0..3
    const int wn   = wave & 1;    // 0..1
    const int bm   = blockIdx.y * BM;
    const int bn   = blockIdx.x * BN;
    const int ln   = lane & 15;   // N / M-fragment lane
    const int lh   = lane >> 4;   // half-wave select

    f32x8 acc[2][4];
#pragma unroll
    for (int mt = 0; mt < 2; mt++)
#pragma unroll
        for (int nt = 0; nt < 4; nt++) { f32x8 z = {}; acc[mt][nt] = z; }

    const int nk = K / BK;
    // prologue: post tile 0 into buffer 0
    if (wave == 0) {
        tdm_load_tile_2d((unsigned)(uintptr_t)&As[0][0], A + (size_t)bm * K,
                         (unsigned)K, (unsigned)M, (unsigned)K, BK, BM, TDM_PAD_40);
        tdm_load_tile_2d((unsigned)(uintptr_t)&Bs[0][0], B + bn,
                         (unsigned)N, (unsigned)K, (unsigned)N, BN, BK, 0u);
    }

    for (int i = 0; i < nk; i++) {
        const int buf = i & 1;
        if (wave == 0) {
            if (i + 1 < nk) {
                int k1 = (i + 1) * BK;
                tdm_load_tile_2d((unsigned)(uintptr_t)&As[buf ^ 1][0],
                                 A + (size_t)bm * K + k1,
                                 (unsigned)K, (unsigned)M, (unsigned)K, BK, BM, TDM_PAD_40);
                tdm_load_tile_2d((unsigned)(uintptr_t)&Bs[buf ^ 1][0],
                                 B + (size_t)k1 * N + bn,
                                 (unsigned)N, (unsigned)K, (unsigned)N, BN, BK, 0u);
                __builtin_amdgcn_s_wait_tensorcnt(2);  // tile i done (in-order)
            } else {
                __builtin_amdgcn_s_wait_tensorcnt(0);
            }
        }
        __syncthreads();  // publish tile i to all waves

        FragAB fa[2], fb[4];
#pragma unroll
        for (int mt = 0; mt < 2; mt++) {
            const bf16_t* base = &As[buf][(wm * 32 + mt * 16 + ln) * LSTA];
#pragma unroll
            for (int v = 0; v < 8; v++) {
                int kk = ((v & 4) ? 16 : 0) + lh * 8 + (v & 3) * 2;  // ISA A-layout
                fa[mt].u[v] = *(const unsigned*)(base + kk);
            }
        }
#pragma unroll
        for (int nt = 0; nt < 4; nt++) {
            const bf16_t* base = &Bs[buf][wn * 64 + nt * 16 + ln];
#pragma unroll
            for (int v = 0; v < 8; v++) {
                int kk = ((v & 4) ? 16 : 0) + lh * 8 + (v & 3) * 2;  // ISA B-layout (matches A)
                unsigned lo = *(const unsigned short*)(base + (size_t)kk * BN);
                unsigned hi = *(const unsigned short*)(base + (size_t)(kk + 1) * BN);
                fb[nt].u[v] = lo | (hi << 16);
            }
        }
#pragma unroll
        for (int mt = 0; mt < 2; mt++)
#pragma unroll
            for (int nt = 0; nt < 4; nt++)
                acc[mt][nt] = wmma_bf16(fa[mt], fb[nt], acc[mt][nt]);
        __syncthreads();  // tile i consumed; buffer may be overwritten next iter
    }

    // --- epilogue: C layout lane holds (m = v + 8*lh, n = ln) ---
#pragma unroll
    for (int mt = 0; mt < 2; mt++)
#pragma unroll
        for (int nt = 0; nt < 4; nt++)
#pragma unroll
            for (int v = 0; v < 8; v++) {
                int r = bm + wm * 32 + mt * 16 + v + 8 * lh;
                int c = bn + wn * 64 + nt * 16 + ln;
                size_t idx = (size_t)r * N + c;
                float val = acc[mt][nt][v];
                if (EPI == 0) {
                    ((float*)OutV)[idx] = val;
                } else if (EPI == 1) {
                    ((float*)OutV)[idx] += val;
                } else {
                    float g = Aux[idx];
                    float sg = g / (1.f + expf(-g));  // silu(x2@W1)
                    ((bf16_t*)OutV)[idx] = (bf16_t)(sg * val);
                }
            }
}

// ---------------------------------------------------------------------------
// Causal flash attention, one head per blockIdx.x, 128 query rows per block,
// one wave per 16-query tile; keys in steps of 32 (two 16x16 score tiles).
// Q,K: bf16 [S][D] (post-RoPE). Vt: bf16 [D][S]. O: bf16 [S][D].
// ---------------------------------------------------------------------------
__global__ __launch_bounds__(256) void attn_kernel(
    const bf16_t* __restrict__ Q, const bf16_t* __restrict__ Kc,
    const bf16_t* __restrict__ Vt, bf16_t* __restrict__ O,
    int S, int D, float scale) {
    __shared__ bf16_t pbuf[8][16 * 36];  // per-wave prob bounce buffer (padded rows)

    const int tid  = threadIdx.x;
    const int lane = tid & 31;
    const int wave = tid >> 5;
    const int ln   = lane & 15;
    const int lh   = lane >> 4;
    const int h    = blockIdx.x;
    const int qb   = blockIdx.y * 128 + wave * 16;
    const int hd0  = h * 64;

    // Q fragments (16 rows x 64 dims = two 16x32 A-frags), loaded once
    FragAB aq[2];
#pragma unroll
    for (int ks = 0; ks < 2; ks++) {
        const bf16_t* qrow = Q + (size_t)(qb + ln) * D + hd0 + ks * 32;
#pragma unroll
        for (int v = 0; v < 8; v++) {
            int kk = ((v & 4) ? 16 : 0) + lh * 8 + (v & 3) * 2;
            aq[ks].u[v] = *(const unsigned*)(qrow + kk);
        }
    }

    f32x8 acco[4];
#pragma unroll
    for (int nt = 0; nt < 4; nt++) { f32x8 z = {}; acco[nt] = z; }
    float mrow[8], lrow[8];
#pragma unroll
    for (int v = 0; v < 8; v++) { mrow[v] = -3e38f; lrow[v] = 0.f; }

    const int kend = qb + 16;  // causal: keys < kend
    for (int kb = 0; kb < kend; kb += 32) {
        // ---- scores: two 16x16 tiles over hd=64 (two WMMA K-steps each) ----
        f32x8 sc[2];
#pragma unroll
        for (int t = 0; t < 2; t++) {
            f32x8 z = {};
            sc[t] = z;
#pragma unroll
            for (int ks = 0; ks < 2; ks++) {
                FragAB bk;
                int key = kb + t * 16 + ln;  // S % 32 == 0 => key < S always
                const bf16_t* krow = Kc + (size_t)key * D + hd0 + ks * 32;
#pragma unroll
                for (int v = 0; v < 8; v++) {
                    int kk = ((v & 4) ? 16 : 0) + lh * 8 + (v & 3) * 2;
                    bk.u[v] = *(const unsigned*)(krow + kk);
                }
                sc[t] = wmma_bf16(aq[ks], bk, sc[t]);
            }
        }
        // ---- online softmax (rows split across half-waves per C layout) ----
        float pr[2][8];
#pragma unroll
        for (int v = 0; v < 8; v++) {
            int rrow = qb + v + 8 * lh;
            float s0 = sc[0][v] * scale;
            float s1 = sc[1][v] * scale;
            if (kb + ln > rrow)      s0 = -3e38f;
            if (kb + 16 + ln > rrow) s1 = -3e38f;
            float mx = fmaxf(s0, s1);
#pragma unroll
            for (int o = 1; o < 16; o <<= 1) mx = fmaxf(mx, __shfl_xor(mx, o, 32));
            float mnew = fmaxf(mrow[v], mx);
            float corr = expf(mrow[v] - mnew);
            float p0 = expf(s0 - mnew);
            float p1 = expf(s1 - mnew);
            float ps = p0 + p1;
#pragma unroll
            for (int o = 1; o < 16; o <<= 1) ps += __shfl_xor(ps, o, 32);
            lrow[v] = lrow[v] * corr + ps;
            mrow[v] = mnew;
            pr[0][v] = p0;
            pr[1][v] = p1;
#pragma unroll
            for (int nt = 0; nt < 4; nt++) acco[nt][v] *= corr;
        }
        // ---- C-layout probs -> A-frag layout via per-wave LDS bounce ----
        bf16_t* pb = &pbuf[wave][0];
#pragma unroll
        for (int t = 0; t < 2; t++)
#pragma unroll
            for (int v = 0; v < 8; v++)
                pb[(v + 8 * lh) * 36 + t * 16 + ln] = (bf16_t)pr[t][v];
        asm volatile("s_wait_dscnt 0" ::: "memory");  // same-wave LDS RAW fence
        FragAB ap;
#pragma unroll
        for (int v = 0; v < 8; v++) {
            int kk = ((v & 4) ? 16 : 0) + lh * 8 + (v & 3) * 2;
            ap.u[v] = *(const unsigned*)(pb + ln * 36 + kk);
        }
        // ---- P(16x32) x V(32x64): V-frags from transposed V, pairs contiguous ----
#pragma unroll
        for (int nt = 0; nt < 4; nt++) {
            FragAB bv;
            const bf16_t* vrow = Vt + (size_t)(hd0 + nt * 16 + ln) * S + kb;
#pragma unroll
            for (int v = 0; v < 8; v++) {
                int kk = ((v & 4) ? 16 : 0) + lh * 8 + (v & 3) * 2;
                bv.u[v] = *(const unsigned*)(vrow + kk);
            }
            acco[nt] = wmma_bf16(ap, bv, acco[nt]);
        }
    }

    // ---- finalize: o = acc / l, write bf16 [S][D] ----
#pragma unroll
    for (int nt = 0; nt < 4; nt++)
#pragma unroll
        for (int v = 0; v < 8; v++) {
            int r = qb + v + 8 * lh;
            float o = acco[nt][v] / lrow[v];
            O[(size_t)r * D + hd0 + nt * 16 + ln] = (bf16_t)o;
        }
}

// ---------------------------------------------------------------------------
// Logits GEMV: out[n] = sum_k hn[k] * Wout[k][n]  (M=1, memory-bound -> f32)
// ---------------------------------------------------------------------------
__global__ __launch_bounds__(256) void gemv_out_kernel(const float* __restrict__ hn,
                                                       const float* __restrict__ W,
                                                       float* __restrict__ out,
                                                       int Kd, int V) {
    __shared__ float hs[1024];
    for (int i = threadIdx.x; i < Kd; i += 256) hs[i] = hn[i];
    __syncthreads();
    int n = blockIdx.x * 256 + threadIdx.x;
    if (n < V) {
        float acc = 0.f;
        for (int k = 0; k < Kd; k++) acc += hs[k] * W[(size_t)k * V + n];
        out[n] = acc;
    }
}

// ---------------------------------------------------------------------------
// Host-side orchestration
// ---------------------------------------------------------------------------
extern "C" void kernel_launch(void* const* d_in, const int* in_sizes, int n_in,
                              void* d_out, int out_size, void* d_ws, size_t ws_size,
                              hipStream_t stream) {
    (void)in_sizes; (void)n_in; (void)out_size; (void)ws_size;
    const int S = 2048, D = 1024, L = 4, V = 32000, Hff = 2816, H = 16, HD = 64;

    const int*   tokens   = (const int*)d_in[0];
    const float* tok_emb  = (const float*)d_in[2];
    const float* Wq       = (const float*)d_in[3];
    const float* Wk       = (const float*)d_in[4];
    const float* Wv       = (const float*)d_in[5];
    const float* Wo       = (const float*)d_in[6];
    const float* W1       = (const float*)d_in[7];
    const float* W2       = (const float*)d_in[8];
    const float* W3       = (const float*)d_in[9];
    const float* attn_nw  = (const float*)d_in[10];
    const float* ffn_nw   = (const float*)d_in[11];
    const float* final_nw = (const float*)d_in[12];
    const float* Wout     = (const float*)d_in[13];

    char* ws = (char*)d_ws;
    size_t off = 0;
    auto alloc = [&](size_t bytes) -> char* {
        off = (off + 255) & ~(size_t)255;
        char* p = ws + off;
        off += bytes;
        return p;
    };
    float*  h    = (float*)alloc((size_t)S * D * 4);
    bf16_t* xn   = (bf16_t*)alloc((size_t)S * D * 2);
    float*  qf   = (float*)alloc((size_t)S * D * 4);
    float*  kf   = (float*)alloc((size_t)S * D * 4);
    float*  vf   = (float*)alloc((size_t)S * D * 4);
    bf16_t* qb   = (bf16_t*)alloc((size_t)S * D * 2);
    bf16_t* kb   = (bf16_t*)alloc((size_t)S * D * 2);
    bf16_t* vt   = (bf16_t*)alloc((size_t)S * D * 2);
    bf16_t* ob   = (bf16_t*)alloc((size_t)S * D * 2);
    float*  g1   = (float*)alloc((size_t)S * Hff * 4);
    bf16_t* ga   = (bf16_t*)alloc((size_t)S * Hff * 2);
    float*  hn   = (float*)alloc((size_t)D * 4);
    bf16_t* wq_b = (bf16_t*)alloc((size_t)D * D * 2);
    bf16_t* wk_b = (bf16_t*)alloc((size_t)D * D * 2);
    bf16_t* wv_b = (bf16_t*)alloc((size_t)D * D * 2);
    bf16_t* wo_b = (bf16_t*)alloc((size_t)D * D * 2);
    bf16_t* w1_b = (bf16_t*)alloc((size_t)D * Hff * 2);
    bf16_t* w3_b = (bf16_t*)alloc((size_t)D * Hff * 2);
    bf16_t* w2_b = (bf16_t*)alloc((size_t)Hff * D * 2);

    auto cvt = [&](const float* src, bf16_t* dst, size_t n) {
        f2bf_kernel<<<(unsigned)((n + 255) / 256), 256, 0, stream>>>(src, dst, n);
    };

    // embedding
    embed_kernel<<<(S * D + 255) / 256, 256, 0, stream>>>(tokens, tok_emb, h, S, D);

    const dim3 gD(D / 128, S / 128);     // N=1024 GEMMs
    const dim3 gF(Hff / 128, S / 128);   // N=2816 GEMMs

    for (int l = 0; l < L; l++) {
        cvt(Wq + (size_t)l * D * D,  wq_b, (size_t)D * D);
        cvt(Wk + (size_t)l * D * D,  wk_b, (size_t)D * D);
        cvt(Wv + (size_t)l * D * D,  wv_b, (size_t)D * D);
        cvt(Wo + (size_t)l * D * D,  wo_b, (size_t)D * D);
        cvt(W1 + (size_t)l * D * Hff, w1_b, (size_t)D * Hff);
        cvt(W3 + (size_t)l * D * Hff, w3_b, (size_t)D * Hff);
        cvt(W2 + (size_t)l * Hff * D, w2_b, (size_t)Hff * D);

        // attention block
        rmsnorm_bf16_kernel<<<S, 256, 0, stream>>>(h, attn_nw + (size_t)l * D, xn, D);
        gemm_bf16_kernel<0><<<gD, 256, 0, stream>>>(xn, wq_b, qf, nullptr, S, D, D);
        gemm_bf16_kernel<0><<<gD, 256, 0, stream>>>(xn, wk_b, kf, nullptr, S, D, D);
        gemm_bf16_kernel<0><<<gD, 256, 0, stream>>>(xn, wv_b, vf, nullptr, S, D, D);
        rope_bf16_kernel<<<(S * D / 2 + 255) / 256, 256, 0, stream>>>(qf, qb, S, D, HD);
        rope_bf16_kernel<<<(S * D / 2 + 255) / 256, 256, 0, stream>>>(kf, kb, S, D, HD);
        vtrans_kernel<<<(S * D + 255) / 256, 256, 0, stream>>>(vf, vt, S, D);
        attn_kernel<<<dim3(H, S / 128), 256, 0, stream>>>(qb, kb, vt, ob, S, D, 0.125f);
        gemm_bf16_kernel<1><<<gD, 256, 0, stream>>>(ob, wo_b, h, nullptr, S, D, D);

        // FFN block
        rmsnorm_bf16_kernel<<<S, 256, 0, stream>>>(h, ffn_nw + (size_t)l * D, xn, D);
        gemm_bf16_kernel<0><<<gF, 256, 0, stream>>>(xn, w1_b, g1, nullptr, S, Hff, D);
        gemm_bf16_kernel<2><<<gF, 256, 0, stream>>>(xn, w3_b, ga, g1, S, Hff, D);
        gemm_bf16_kernel<1><<<gD, 256, 0, stream>>>(ga, w2_b, h, nullptr, S, D, Hff);
    }

    final_norm_kernel<<<1, 256, 0, stream>>>(h, final_nw, hn, D, S - 1);
    gemv_out_kernel<<<(V + 255) / 256, 256, 0, stream>>>(hn, Wout, (float*)d_out, D, V);
}
